// Attention_5111011082487
// MI455X (gfx1250) — matrix operand order repaired and checked by reading, not compiled, not run on hardware
//
#include <hip/hip_runtime.h>
#include <hip/hip_bf16.h>
#include <math.h>

// ---- problem constants (match reference) ----
#define D_MODEL 768
#define N_HEADS 12
#define D_HEAD  64
#define BATCH   2
#define SEQ     2048

typedef __attribute__((ext_vector_type(16))) _Float16 v16h;
typedef __attribute__((ext_vector_type(8)))  float    v8f;
typedef __attribute__((ext_vector_type(4)))  float    f4;
typedef __attribute__((ext_vector_type(4)))  _Float16 h4;
typedef __attribute__((ext_vector_type(2)))  _Float16 h2;

// D = A(16x32 f16) * B(32x16 f16) + C(16x16 f32)
__device__ __forceinline__ v8f wmma_f16(v16h a, v16h b, v8f c) {
    return __builtin_amdgcn_wmma_f32_16x16x32_f16(
        /*neg_a=*/false, a, /*neg_b=*/false, b,
        /*c_mod=*/(short)0, c, /*reuse_a=*/false, /*reuse_b=*/false);
}

// A-fragment: 16x32 f16 tile, row-major, row stride `ld` (halfs).
// Lane L: m = L&15; halfs 0..15 <- K = kbase..kbase+15 (kbase=(L>>4)*16)
//                   (two contiguous 16B runs -> 2x b128)
__device__ __forceinline__ v16h frag_a_from(const _Float16* base, int ld) {
    const int lane = threadIdx.x & 31;
    const int m  = lane & 15;
    const int kb = (lane >> 4) * 16;
    const _Float16* r = base + (size_t)m * ld;
    v16h a;
#pragma unroll
    for (int i = 0; i < 8; ++i) {
        a[i]     = r[kb + i];
        a[8 + i] = r[kb + 8 + i];
    }
    return a;
}

// B-fragment: logical B is 32(K) x 16(N), read from B^T stored [n][k],
// row stride `ld` (halfs); each lane reads 16 contiguous halfs (2x b128).
__device__ __forceinline__ v16h frag_b_fromT(const _Float16* baseT, int ld) {
    const int lane = threadIdx.x & 31;
    const int n  = lane & 15;
    const int kb = (lane >> 4) * 16;
    const _Float16* r = baseT + (size_t)n * ld + kb;
    v16h b;
#pragma unroll
    for (int j = 0; j < 16; ++j) b[j] = r[j];
    return b;
}

// ---------------------------------------------------------------------------
// Prep A: x f32 -> f16, straight copy (vector f4 -> h4).
// ---------------------------------------------------------------------------
__global__ void __launch_bounds__(256)
cvt_x_kernel(const float* __restrict__ x, _Float16* __restrict__ xh) {
    const size_t i = (size_t)blockIdx.x * 256 + threadIdx.x;  // f4 index
    f4 v = ((const f4*)x)[i];
    h4 h;
    h[0] = (_Float16)v[0]; h[1] = (_Float16)v[1];
    h[2] = (_Float16)v[2]; h[3] = (_Float16)v[3];
    ((h4*)xh)[i] = h;
}

// ---------------------------------------------------------------------------
// Prep B: masked weights -> f16, transposed.
//   which=0..2: W_{Q,K,V} [h][d][e] -> T [h][e][d]
//   which=3   : W_O flat [k=768][n=768] -> T [n][k]
// Each matrix has exactly 589824 elements = 2304 blocks x 256 threads.
// ---------------------------------------------------------------------------
__global__ void __launch_bounds__(256)
prep_w_kernel(const float* __restrict__ WQ, const float* __restrict__ mQ,
              const float* __restrict__ WK, const float* __restrict__ mK,
              const float* __restrict__ WV, const float* __restrict__ mV,
              const float* __restrict__ WO, const float* __restrict__ mO,
              _Float16* __restrict__ WqT, _Float16* __restrict__ WkT,
              _Float16* __restrict__ WvT, _Float16* __restrict__ WoT)
{
    const int which = blockIdx.y;
    const int o = blockIdx.x * 256 + threadIdx.x;

    const float* W; const float* M; _Float16* T;
    if (which == 0)      { W = WQ; M = mQ; T = WqT; }
    else if (which == 1) { W = WK; M = mK; T = WkT; }
    else if (which == 2) { W = WV; M = mV; T = WvT; }
    else                 { W = WO; M = mO; T = WoT; }

    size_t in;
    if (which < 3) {
        const int h = o / (D_HEAD * D_MODEL);
        const int r = o % (D_HEAD * D_MODEL);
        const int e = r / D_MODEL;
        const int d = r % D_MODEL;
        in = (size_t)h * D_MODEL * D_HEAD + (size_t)d * D_HEAD + e;
    } else {
        const int nn = o / D_MODEL;
        const int k  = o % D_MODEL;
        in = (size_t)k * D_MODEL + nn;
    }
    T[o] = (_Float16)(W[in] * M[in]);
}

// ---------------------------------------------------------------------------
// Kernel 1: QKV projection from pre-converted operands. One wave computes a
// 32x64 output tile, K=768 in 24 steps of 32 (8 WMMA / step), software-
// pipelined: B-frags batched per step, next step's A-frags in flight.
//   blockIdx.x: 32-row tile over B*S ; y: head ; z: 0=Q,1=K,2=V
// Q stored pre-scaled by 1/8; V stored transposed [b,h,e,s].
// ---------------------------------------------------------------------------
__global__ void __launch_bounds__(32)
qkv_gemm_kernel(const _Float16* __restrict__ xh,
                const _Float16* __restrict__ WqT, const _Float16* __restrict__ WkT,
                const _Float16* __restrict__ WvT,
                const float* __restrict__ bQ, const float* __restrict__ bK,
                const float* __restrict__ bV,
                _Float16* __restrict__ Qh, _Float16* __restrict__ Kh,
                _Float16* __restrict__ Vt)
{
    const int lane  = threadIdx.x;
    const int tile  = blockIdx.x;
    const int h     = blockIdx.y;
    const int which = blockIdx.z;
    const int b     = tile / (SEQ / 32);
    const int srow  = (tile % (SEQ / 32)) * 32;

    const _Float16* WT; const float* bias;
    if (which == 0)      { WT = WqT; bias = bQ; }
    else if (which == 1) { WT = WkT; bias = bK; }
    else                 { WT = WvT; bias = bV; }

    const _Float16* xrow = xh + (size_t)(b * SEQ + srow) * D_MODEL;
    const _Float16* Wh   = WT + (size_t)h * D_HEAD * D_MODEL;

    v8f zero = {};
    v8f acc[2][4];
#pragma unroll
    for (int rt = 0; rt < 2; ++rt)
#pragma unroll
        for (int j = 0; j < 4; ++j) acc[rt][j] = zero;

    // prologue: first A-fragments
    v16h a0 = frag_a_from(xrow, D_MODEL);
    v16h a1 = frag_a_from(xrow + (size_t)16 * D_MODEL, D_MODEL);

    for (int kk = 0; kk < D_MODEL; kk += 32) {
        // batch all 4 B-fragments for this step (8 outstanding b128 loads)
        v16h bf[4];
#pragma unroll
        for (int j = 0; j < 4; ++j)
            bf[j] = frag_b_fromT(Wh + (size_t)(j * 16) * D_MODEL + kk, D_MODEL);

        // keep current A, start next step's A-fragment loads
        const v16h a0c = a0, a1c = a1;
        if (kk + 32 < D_MODEL) {
            a0 = frag_a_from(xrow + kk + 32, D_MODEL);
            a1 = frag_a_from(xrow + (size_t)16 * D_MODEL + kk + 32, D_MODEL);
            __builtin_prefetch(xrow + kk + 64, 0, 1);   // global_prefetch_b8
        }

#pragma unroll
        for (int j = 0; j < 4; ++j) {
            acc[0][j] = wmma_f16(a0c, bf[j], acc[0][j]);
            acc[1][j] = wmma_f16(a1c, bf[j], acc[1][j]);
        }
    }

    const int n  = lane & 15;
    const int rb = (lane >> 4) * 8;
    const size_t bh = (size_t)(b * N_HEADS + h);
#pragma unroll
    for (int rt = 0; rt < 2; ++rt)
#pragma unroll
        for (int j = 0; j < 4; ++j) {
            const int e = j * 16 + n;
            const float bv = bias[h * D_HEAD + e];
            if (which == 2) {
                // V transposed: contiguous along r -> pack half2 stores
                _Float16* vp = Vt + (bh * D_HEAD + e) * SEQ + srow + rt * 16 + rb;
#pragma unroll
                for (int r = 0; r < 8; r += 2) {
                    h2 hv;
                    hv[0] = (_Float16)(acc[rt][j][r] + bv);
                    hv[1] = (_Float16)(acc[rt][j][r + 1] + bv);
                    *(h2*)(vp + r) = hv;
                }
            } else {
                const float sc = (which == 0) ? 0.125f : 1.0f;  // fold 1/sqrt(64) into Q
                _Float16* dst = (which == 0) ? Qh : Kh;
#pragma unroll
                for (int r = 0; r < 8; ++r) {
                    const int m = rt * 16 + rb + r;
                    dst[(bh * SEQ + srow + m) * D_HEAD + e] =
                        (_Float16)((acc[rt][j][r] + bv) * sc);
                }
            }
        }
}

// ---------------------------------------------------------------------------
// Kernel 2: causal flash attention. One wave handles 32 query rows of one
// (b,h): 16 WMMAs per 32-key block; softmax uses all 32 lanes (lane == row).
// ---------------------------------------------------------------------------
__global__ void __launch_bounds__(32)
flash_attn_kernel(const _Float16* __restrict__ Qh, const _Float16* __restrict__ Kh,
                  const _Float16* __restrict__ Vt, _Float16* __restrict__ Zh)
{
    __shared__ __align__(16) float    sc[32 * 32];   // raw scores tile
    __shared__ __align__(16) _Float16 pt[32 * 32];   // P tile (A-layout source)
    __shared__ float brA[32];                        // per-row alpha
    __shared__ float brL[32];                        // per-row 1/l

    const int lane  = threadIdx.x;
    const int qbase = blockIdx.x * 32;
    const int h     = blockIdx.y;
    const int b     = blockIdx.z;
    const size_t bh = (size_t)(b * N_HEADS + h);

    const _Float16* Qp = Qh + (bh * SEQ + qbase) * D_HEAD;
    const _Float16* Kp = Kh + bh * SEQ * D_HEAD;
    const _Float16* Vp = Vt + bh * (size_t)D_HEAD * SEQ;

    // Q fragments: 2 row-tiles x 2 e-steps
    v16h qa[2][2];
#pragma unroll
    for (int rt = 0; rt < 2; ++rt) {
        qa[rt][0] = frag_a_from(Qp + (size_t)(rt * 16) * D_HEAD, D_HEAD);
        qa[rt][1] = frag_a_from(Qp + (size_t)(rt * 16) * D_HEAD + 32, D_HEAD);
    }

    v8f zero = {};
    v8f zacc[2][4];
#pragma unroll
    for (int rt = 0; rt < 2; ++rt)
#pragma unroll
        for (int j = 0; j < 4; ++j) zacc[rt][j] = zero;

    float mrow = -1e30f, lrow = 0.0f;                // lane == row (0..31)
    const int n  = lane & 15;
    const int rb = (lane >> 4) * 8;

    for (int kb = 0; kb <= qbase; kb += 32) {        // kb mult of 32, <= qbase+31
        // ---- K fragments (8 outstanding b128 pairs), then 8 score WMMAs ----
        v16h kf[2][2];
#pragma unroll
        for (int ct = 0; ct < 2; ++ct) {
            kf[ct][0] = frag_b_fromT(Kp + (size_t)(kb + ct * 16) * D_HEAD, D_HEAD);
            kf[ct][1] = frag_b_fromT(Kp + (size_t)(kb + ct * 16) * D_HEAD + 32, D_HEAD);
        }
        // batch V fragments early too: independent of scores, hides latency
        v16h vb[4];
#pragma unroll
        for (int j = 0; j < 4; ++j)
            vb[j] = frag_b_fromT(Vp + (size_t)(j * 16) * SEQ + kb, SEQ);

        v8f s[2][2];
#pragma unroll
        for (int rt = 0; rt < 2; ++rt)
#pragma unroll
            for (int ct = 0; ct < 2; ++ct) s[rt][ct] = zero;
#pragma unroll
        for (int es = 0; es < 2; ++es)
#pragma unroll
            for (int rt = 0; rt < 2; ++rt)
#pragma unroll
                for (int ct = 0; ct < 2; ++ct)
                    s[rt][ct] = wmma_f16(qa[rt][es], kf[ct][es], s[rt][ct]);

        // ---- causal mask in C layout, spill to LDS ----
#pragma unroll
        for (int rt = 0; rt < 2; ++rt)
#pragma unroll
            for (int ct = 0; ct < 2; ++ct)
#pragma unroll
                for (int r = 0; r < 8; ++r) {
                    const int m   = rt * 16 + rb + r;
                    const int col = ct * 16 + n;
                    const float v = ((kb + col) <= (qbase + m)) ? s[rt][ct][r] : -1e30f;
                    sc[m * 32 + col] = v;
                }
        __syncthreads();

        // ---- online softmax: lane owns row `lane` ----
        float rmax = mrow;
#pragma unroll
        for (int k2 = 0; k2 < 32; k2 += 4) {
            const f4 sv = *(const f4*)(sc + lane * 32 + k2);
            rmax = fmaxf(rmax, fmaxf(fmaxf(sv[0], sv[1]), fmaxf(sv[2], sv[3])));
        }
        const float alpha = __expf(mrow - rmax);
        float psum = 0.0f;
#pragma unroll
        for (int k2 = 0; k2 < 32; k2 += 4) {
            const f4 sv = *(const f4*)(sc + lane * 32 + k2);
            const float p0 = __expf(sv[0] - rmax), p1 = __expf(sv[1] - rmax);
            const float p2 = __expf(sv[2] - rmax), p3 = __expf(sv[3] - rmax);
            h4 hv;
            hv[0] = (_Float16)p0; hv[1] = (_Float16)p1;
            hv[2] = (_Float16)p2; hv[3] = (_Float16)p3;
            *(h4*)(pt + lane * 32 + k2) = hv;        // ds_store_b64
            psum += (p0 + p1) + (p2 + p3);
        }
        mrow = rmax;
        lrow = lrow * alpha + psum;
        brA[lane] = alpha;
        __syncthreads();

        // ---- rescale accumulators by alpha ----
#pragma unroll
        for (int rt = 0; rt < 2; ++rt)
#pragma unroll
            for (int r = 0; r < 8; ++r) {
                const float a = brA[rt * 16 + rb + r];
#pragma unroll
                for (int j = 0; j < 4; ++j) zacc[rt][j][r] *= a;
            }

        // ---- z += P @ V (V frags already in flight) ----
        v16h pa[2];
#pragma unroll
        for (int rt = 0; rt < 2; ++rt) pa[rt] = frag_a_from(pt + rt * 16 * 32, 32);
#pragma unroll
        for (int j = 0; j < 4; ++j) {
            zacc[0][j] = wmma_f16(pa[0], vb[j], zacc[0][j]);
            zacc[1][j] = wmma_f16(pa[1], vb[j], zacc[1][j]);
        }
        __syncthreads();
    }

    // ---- finalize: z /= l, store f16 z as [b, s, h*64+e] ----
    brL[lane] = 1.0f / lrow;
    __syncthreads();

    _Float16* Zp = Zh + ((size_t)b * SEQ + qbase) * D_MODEL + h * D_HEAD;
#pragma unroll
    for (int rt = 0; rt < 2; ++rt)
#pragma unroll
        for (int j = 0; j < 4; ++j)
#pragma unroll
            for (int r = 0; r < 8; ++r) {
                const int m = rt * 16 + rb + r;
                Zp[(size_t)m * D_MODEL + j * 16 + n] = (_Float16)(zacc[rt][j][r] * brL[m]);
            }
}

// ---------------------------------------------------------------------------
// Kernel 3: output projection  out = z @ WoT^T + b_O  (f32 out), 32x64 tiles,
// software-pipelined like kernel 1.
// ---------------------------------------------------------------------------
__global__ void __launch_bounds__(32)
out_proj_kernel(const _Float16* __restrict__ Zh, const _Float16* __restrict__ WoT,
                const float* __restrict__ bO, float* __restrict__ out)
{
    const int lane  = threadIdx.x;
    const int srow  = blockIdx.x * 32;     // over flattened B*S
    const int nbase = blockIdx.y * 64;     // output column block

    const _Float16* Zp = Zh  + (size_t)srow * D_MODEL;
    const _Float16* Wn = WoT + (size_t)nbase * D_MODEL;

    v8f zero = {};
    v8f acc[2][4];
#pragma unroll
    for (int rt = 0; rt < 2; ++rt)
#pragma unroll
        for (int j = 0; j < 4; ++j) acc[rt][j] = zero;

    v16h a0 = frag_a_from(Zp, D_MODEL);
    v16h a1 = frag_a_from(Zp + (size_t)16 * D_MODEL, D_MODEL);

    for (int kk = 0; kk < D_MODEL; kk += 32) {
        v16h bf[4];
#pragma unroll
        for (int j = 0; j < 4; ++j)
            bf[j] = frag_b_fromT(Wn + (size_t)(j * 16) * D_MODEL + kk, D_MODEL);

        const v16h a0c = a0, a1c = a1;
        if (kk + 32 < D_MODEL) {
            a0 = frag_a_from(Zp + kk + 32, D_MODEL);
            a1 = frag_a_from(Zp + (size_t)16 * D_MODEL + kk + 32, D_MODEL);
            __builtin_prefetch(Zp + kk + 64, 0, 1);
        }

#pragma unroll
        for (int j = 0; j < 4; ++j) {
            acc[0][j] = wmma_f16(a0c, bf[j], acc[0][j]);
            acc[1][j] = wmma_f16(a1c, bf[j], acc[1][j]);
        }
    }

    const int n  = lane & 15;
    const int rb = (lane >> 4) * 8;
#pragma unroll
    for (int rt = 0; rt < 2; ++rt)
#pragma unroll
        for (int j = 0; j < 4; ++j) {
            const float bv = bO[nbase + j * 16 + n];
#pragma unroll
            for (int r = 0; r < 8; ++r) {
                const int m = rt * 16 + rb + r;
                out[(size_t)(srow + m) * D_MODEL + nbase + j * 16 + n] = acc[rt][j][r] + bv;
            }
        }
}

// ---------------------------------------------------------------------------
extern "C" void kernel_launch(void* const* d_in, const int* in_sizes, int n_in,
                              void* d_out, int out_size, void* d_ws, size_t ws_size,
                              hipStream_t stream) {
    (void)in_sizes; (void)n_in; (void)out_size; (void)ws_size;
    const float* x  = (const float*)d_in[0];
    const float* WQ = (const float*)d_in[1];
    const float* bQ = (const float*)d_in[2];
    const float* WK = (const float*)d_in[3];
    const float* bK = (const float*)d_in[4];
    const float* WV = (const float*)d_in[5];
    const float* bV = (const float*)d_in[6];
    const float* WO = (const float*)d_in[7];
    const float* bO = (const float*)d_in[8];
    const float* mQ = (const float*)d_in[9];
    const float* mK = (const float*)d_in[10];
    const float* mV = (const float*)d_in[11];
    const float* mO = (const float*)d_in[12];

    // workspace layout (halfs)
    const size_t SQ   = (size_t)BATCH * SEQ * D_MODEL;          // 3,145,728
    const size_t PH   = (size_t)BATCH * N_HEADS * SEQ * D_HEAD; // 3,145,728
    const size_t WSZ  = (size_t)N_HEADS * D_HEAD * D_MODEL;     //   589,824
    _Float16* xh  = (_Float16*)d_ws;
    _Float16* Qh  = xh  + SQ;
    _Float16* Kh  = Qh  + PH;
    _Float16* Vt  = Kh  + PH;
    _Float16* Zh  = Vt  + PH;
    _Float16* WqT = Zh  + SQ;
    _Float16* WkT = WqT + WSZ;
    _Float16* WvT = WkT + WSZ;
    _Float16* WoT = WvT + WSZ;                                  // total ~36 MB

    // prep: x -> f16 ; weights -> masked f16 transposed
    cvt_x_kernel<<<dim3((unsigned)(SQ / 4 / 256)), 256, 0, stream>>>(x, xh);
    prep_w_kernel<<<dim3(2304, 4), 256, 0, stream>>>(WQ, mQ, WK, mK, WV, mV, WO, mO,
                                                     WqT, WkT, WvT, WoT);

    dim3 g1(BATCH * SEQ / 32, N_HEADS, 3);
    qkv_gemm_kernel<<<g1, 32, 0, stream>>>(xh, WqT, WkT, WvT, bQ, bK, bV, Qh, Kh, Vt);

    dim3 g2(SEQ / 32, N_HEADS, BATCH);
    flash_attn_kernel<<<g2, 32, 0, stream>>>(Qh, Kh, Vt, Zh);

    dim3 g3(BATCH * SEQ / 32, D_MODEL / 64);
    out_proj_kernel<<<g3, 32, 0, stream>>>(Zh, WoT, bO, (float*)d_out);
}
